// InteractionLayer_72602127171983
// MI455X (gfx1250) — compile-verified, matrix operand order checked by reading, not run on hardware
//
#include <hip/hip_runtime.h>

#define LN2F   0.69314718055994530942f
#define LOG2EF 1.44269504088896340736f

typedef __attribute__((ext_vector_type(16))) __bf16 v16bf;
typedef __attribute__((ext_vector_type(8)))  float  v8f;

// ---------------------------------------------------------------------------
// Shifted softplus via raw CDNA5 transcendentals:
//   ssp(x) = relu(x) + ln2*(log2(1 + 2^(-|x|*log2e)) - 1)
__device__ __forceinline__ float sspf(float x) {
    float t    = __builtin_amdgcn_exp2f(-LOG2EF * fabsf(x));   // v_exp_f32
    float l    = __builtin_amdgcn_logf(1.0f + t);              // v_log_f32
    float relu = __builtin_amdgcn_fmed3f(x, 0.0f, __builtin_inff());
    return fmaf(LN2F, l - 1.0f, relu);
}

// K-offset of element pair v (0..7) of a 16x32 bf16 WMMA A/B fragment
// (CDNA5 ISA 7.12.2: lanes 0-15 hold K {0..7,16..23}, lanes 16-31 {8..15,24..31}).
__device__ __forceinline__ int a_kbase(int lane, int v) {
    int hi = (lane & 16) ? 8 : 0;
    return (v < 4) ? (hi + 2 * v) : (16 + hi + 2 * (v - 4));
}

// Load a 32x16 bf16 B-fragment of row-major W[64][64] for col tile `ntile`,
// K-step `kk`.
__device__ __forceinline__ v16bf load_w_frag(const float* __restrict__ W,
                                             int ntile, int kk, int lane) {
    v16bf b;
    const int n = ntile * 16 + (lane & 15);
#pragma unroll
    for (int v = 0; v < 8; ++v) {
        const int k = kk * 32 + a_kbase(lane, v);
        b[2 * v]     = (__bf16)W[k * 64 + n];
        b[2 * v + 1] = (__bf16)W[(k + 1) * 64 + n];
    }
    return b;
}

// ---------------------------------------------------------------------------
__global__ void zero_kernel(float* __restrict__ p, int n) {
    int i = blockIdx.x * blockDim.x + threadIdx.x;
    if (i < n) p[i] = 0.0f;
}

// ---------------------------------------------------------------------------
// Y[R,64] = act(X[R,64] @ W[64,64] + bias), one 16-row M-tile per wave.
template <int ACT>
__global__ __launch_bounds__(128) void gemm64_kernel(
    const float* __restrict__ X, const float* __restrict__ W,
    const float* __restrict__ bias, float* __restrict__ Y)
{
    const int tid  = threadIdx.x;
    const int lane = tid & 31;
    const int wave = tid >> 5;
    const int ml   = lane & 15;
    const int m0   = (blockIdx.x * 4 + wave) * 16;

    v16bf wf[4][2];
#pragma unroll
    for (int n = 0; n < 4; ++n)
#pragma unroll
        for (int kk = 0; kk < 2; ++kk)
            wf[n][kk] = load_w_frag(W, n, kk, lane);

    float bv[4];
#pragma unroll
    for (int n = 0; n < 4; ++n) bv[n] = bias[n * 16 + ml];

    // A fragments straight from global (rows of X are contiguous 64-float).
    v16bf a[2];
#pragma unroll
    for (int kk = 0; kk < 2; ++kk)
#pragma unroll
        for (int v = 0; v < 8; ++v) {
            const int k = kk * 32 + a_kbase(lane, v);
            a[kk][2 * v]     = (__bf16)X[(size_t)(m0 + ml) * 64 + k];
            a[kk][2 * v + 1] = (__bf16)X[(size_t)(m0 + ml) * 64 + k + 1];
        }

#pragma unroll
    for (int n = 0; n < 4; ++n) {
        v8f c;
#pragma unroll
        for (int g = 0; g < 8; ++g) c[g] = bv[n];
        c = __builtin_amdgcn_wmma_f32_16x16x32_bf16(false, a[0], false, wf[n][0],
                                                    (short)0, c, false, false);
        c = __builtin_amdgcn_wmma_f32_16x16x32_bf16(false, a[1], false, wf[n][1],
                                                    (short)0, c, false, false);
#pragma unroll
        for (int g = 0; g < 8; ++g) {
            const int row = m0 + (lane >> 4) * 8 + g;
            float val = ACT ? sspf(c[g]) : c[g];
            Y[(size_t)row * 64 + n * 16 + ml] = val;
        }
    }
}

// ---------------------------------------------------------------------------
// Fused pairwise filter MLP + neighbor reduction.
//   h2[b,i,:] += sum_j ssp(ssp(|r_i-r_j| @ W1 + b1) @ W2 + b2) * h1[b,j,:]
// Grid: (N/16 i-blocks, N/jPerSlice j-slices, B). Block: 256 = 8 waves.
// Each wave owns 2 rows i; 16 (i,j)-pairs per WMMA M-tile (one i, 16 j's).
__global__ __launch_bounds__(256) void interaction_kernel(
    const float* __restrict__ r,  const float* __restrict__ h1,
    const float* __restrict__ f1w, const float* __restrict__ f1b,
    const float* __restrict__ f2w, const float* __restrict__ f2b,
    float* __restrict__ h2, int N, int jPerSlice)
{
    __shared__ __align__(16) float  sRI[16 * 64];
    __shared__ __align__(16) float  sRJ[16 * 64];
    __shared__ __align__(16) float  sH1[16 * 64];
    __shared__ __align__(16) __bf16 sW[8][16 * 64];  // per-wave layer-1 output

    const int tid  = threadIdx.x;
    const int lane = tid & 31;
    const int wave = tid >> 5;
    const int ml   = lane & 15;
    const int i0   = blockIdx.x * 16;
    const int j0s  = blockIdx.y * jPerSlice;
    const int b    = blockIdx.z;

    // Weight fragments pinned in registers for the whole kernel.
    v16bf w1f[4][2], w2f[4][2];
#pragma unroll
    for (int n = 0; n < 4; ++n)
#pragma unroll
        for (int kk = 0; kk < 2; ++kk) {
            w1f[n][kk] = load_w_frag(f1w, n, kk, lane);
            w2f[n][kk] = load_w_frag(f2w, n, kk, lane);
        }
    float b1v[4], b2v[4];
#pragma unroll
    for (int n = 0; n < 4; ++n) {
        b1v[n] = f1b[n * 16 + ml];
        b2v[n] = f2b[n * 16 + ml];
    }

    // r_i tile: 16 rows x 64 f = one contiguous 4KB block.
    ((float4*)sRI)[tid] = ((const float4*)(r + (size_t)(b * N + i0) * 64))[tid];

    float acc[2][4];
#pragma unroll
    for (int i = 0; i < 2; ++i)
#pragma unroll
        for (int n = 0; n < 4; ++n) acc[i][n] = 0.0f;

    for (int jb = 0; jb < jPerSlice; jb += 16) {
        const int j0 = j0s + jb;
        __syncthreads();
        ((float4*)sRJ)[tid] = ((const float4*)(r  + (size_t)(b * N + j0) * 64))[tid];
        ((float4*)sH1)[tid] = ((const float4*)(h1 + (size_t)(b * N + j0) * 64))[tid];
        if (jb + 16 < jPerSlice) {
            __builtin_prefetch(r  + (size_t)(b * N + j0 + 16) * 64 + tid * 4, 0, 3);
            __builtin_prefetch(h1 + (size_t)(b * N + j0 + 16) * 64 + tid * 4, 0, 3);
        }
        __syncthreads();

#pragma unroll
        for (int i = 0; i < 2; ++i) {
            const int il = wave * 2 + i;

            // Layer-1 A fragments: rows = 16 j's, K = 64 features of |r_j - r_i|.
            // (sRJ reads are i-invariant; compiler CSEs them across the i-loop.)
            v16bf a1[2];
#pragma unroll
            for (int kk = 0; kk < 2; ++kk)
#pragma unroll
                for (int v = 0; v < 8; ++v) {
                    const int k = kk * 32 + a_kbase(lane, v);
                    float d0 = fabsf(sRJ[ml * 64 + k]     - sRI[il * 64 + k]);
                    float d1 = fabsf(sRJ[ml * 64 + k + 1] - sRI[il * 64 + k + 1]);
                    a1[kk][2 * v]     = (__bf16)d0;
                    a1[kk][2 * v + 1] = (__bf16)d1;
                }

            // Layer 1: C = d @ W1 + b1, ssp, stage bf16 tile to LDS (transpose
            // from C layout to A layout goes through memory; same-wave DS ops
            // complete in order, backend inserts the required s_wait_dscnt).
#pragma unroll
            for (int n = 0; n < 4; ++n) {
                v8f c;
#pragma unroll
                for (int g = 0; g < 8; ++g) c[g] = b1v[n];
                c = __builtin_amdgcn_wmma_f32_16x16x32_bf16(false, a1[0], false, w1f[n][0],
                                                            (short)0, c, false, false);
                c = __builtin_amdgcn_wmma_f32_16x16x32_bf16(false, a1[1], false, w1f[n][1],
                                                            (short)0, c, false, false);
#pragma unroll
                for (int g = 0; g < 8; ++g) {
                    const int row = (lane >> 4) * 8 + g;
                    sW[wave][row * 64 + n * 16 + ml] = (__bf16)sspf(c[g]);
                }
            }

            // Layer-2 A fragments from the staged bf16 tile (dword = K pair).
            v16bf a2[2];
#pragma unroll
            for (int kk = 0; kk < 2; ++kk)
#pragma unroll
                for (int v = 0; v < 8; ++v) {
                    const int k = kk * 32 + a_kbase(lane, v);
                    union { unsigned u; __bf16 h[2]; } cv;
                    cv.u = *(const unsigned*)&sW[wave][ml * 64 + k];
                    a2[kk][2 * v]     = cv.h[0];
                    a2[kk][2 * v + 1] = cv.h[1];
                }

            // Layer 2 + in-register neighbor reduction against h1 (h1 fragment
            // in C layout, loaded just before use to keep live ranges short).
#pragma unroll
            for (int n = 0; n < 4; ++n) {
                v8f c;
#pragma unroll
                for (int g = 0; g < 8; ++g) c[g] = b2v[n];
                c = __builtin_amdgcn_wmma_f32_16x16x32_bf16(false, a2[0], false, w2f[n][0],
                                                            (short)0, c, false, false);
                c = __builtin_amdgcn_wmma_f32_16x16x32_bf16(false, a2[1], false, w2f[n][1],
                                                            (short)0, c, false, false);
                float p = 0.0f;
#pragma unroll
                for (int g = 0; g < 8; ++g) {
                    float hv = sH1[((lane >> 4) * 8 + g) * 64 + n * 16 + ml];
                    p += sspf(c[g]) * hv;
                }
                p += __shfl_xor(p, 16, 32);   // fold the two row-halves
                acc[i][n] += p;
            }
        }
    }

    // Accumulate partial neighbor sums for this j-slice into h2.
    if (lane < 16) {
#pragma unroll
        for (int i = 0; i < 2; ++i) {
            const int il = wave * 2 + i;
            float* dst = h2 + (size_t)(b * N + i0 + il) * 64;
#pragma unroll
            for (int n = 0; n < 4; ++n)
                __hip_atomic_fetch_add(dst + n * 16 + ml, acc[i][n],
                                       __ATOMIC_RELAXED, __HIP_MEMORY_SCOPE_AGENT);
        }
    }
}

// ---------------------------------------------------------------------------
extern "C" void kernel_launch(void* const* d_in, const int* in_sizes, int n_in,
                              void* d_out, int out_size, void* d_ws, size_t ws_size,
                              hipStream_t stream) {
    (void)in_sizes; (void)n_in; (void)out_size; (void)ws_size;
    const int B = 2, N = 1024, F = 64;

    const float* x    = (const float*)d_in[0];
    const float* r    = (const float*)d_in[1];
    const float* aw1w = (const float*)d_in[2];
    const float* aw1b = (const float*)d_in[3];
    const float* aw2w = (const float*)d_in[4];
    const float* aw2b = (const float*)d_in[5];
    const float* aw3w = (const float*)d_in[6];
    const float* aw3b = (const float*)d_in[7];
    const float* f1w  = (const float*)d_in[8];
    const float* f1b  = (const float*)d_in[9];
    const float* f2w  = (const float*)d_in[10];
    const float* f2b  = (const float*)d_in[11];
    float* out = (float*)d_out;

    float* h1 = (float*)d_ws;                // [B,N,F] = 512 KB
    float* h2 = h1 + (size_t)B * N * F;      // [B,N,F]
    float* h3 = h2 + (size_t)B * N * F;      // [B,N,F]

    const int R = B * N;                     // 2048 rows

    zero_kernel<<<(B * N * F + 255) / 256, 256, 0, stream>>>(h2, B * N * F);

    // h1 = x @ aw1 + b1
    gemm64_kernel<0><<<R / 64, 128, 0, stream>>>(x, aw1w, aw1b, h1);

    // h2 = sum_j ssp(ssp(|ri-rj| @ f1 + b1) @ f2 + b2) * h1[j]
    dim3 gB(N / 16, 4, B);                   // 64 i-blocks x 4 j-slices x B
    interaction_kernel<<<gB, 256, 0, stream>>>(r, h1, f1w, f1b, f2w, f2b, h2,
                                               N, N / 4);

    // h3 = ssp(h2 @ aw2 + b2); out = h3 @ aw3 + b3
    gemm64_kernel<1><<<R / 64, 128, 0, stream>>>(h2, aw2w, aw2b, h3);
    gemm64_kernel<0><<<R / 64, 128, 0, stream>>>(h3, aw3w, aw3b, out);
}
